// CrossAttention_44487271252317
// MI455X (gfx1250) — matrix operand order repaired and checked by reading, not compiled, not run on hardware
//
#include <hip/hip_runtime.h>
#include <stdint.h>

// ---------------------------------------------------------------------------
// CrossAttention on MI455X (gfx1250, wave32, WMMA f32_16x16x32_f16)
//
// q-proj -> kv-proj -> flash-attention -> out-proj. fp16 data / fp32 accum.
// GEMM: 64x256 block tile, 8 waves x (32x64) wave tiles, operands staged in
// LDS (coalesced b128 global loads, aligned 16B ds loads for A fragments),
// next-tile global_prefetch. Flash-attn: block-shared K/V chunks in LDS,
// V staged via async global->LDS b128 (ASYNCcnt), online softmax in regs.
//
// WMMA fragment layouts (CDNA5 ISA 7.12.2):
//   A (16x32 f16): lane lid(0..15)=row M; halves 0..7 -> K=8*lg+i,
//                  halves 8..15 -> K=8*lg+16+(i-8)
//   B (32x16 f16): lane lid=column N; halves same K order as A
//   C/D (16x16 f32): reg r -> M=r+8*lg ; lane -> N=lid
// ---------------------------------------------------------------------------

typedef __attribute__((ext_vector_type(16))) _Float16 v16h;
typedef __attribute__((ext_vector_type(8)))  _Float16 v8h;
typedef __attribute__((ext_vector_type(4)))  _Float16 v4h;
typedef __attribute__((ext_vector_type(4)))  float    v4f;
typedef __attribute__((ext_vector_type(8)))  float    v8f;

#define WMMA_F16(a, b, c) \
  __builtin_amdgcn_wmma_f32_16x16x32_f16(false, (a), false, (b), (short)0, (c), false, false)

static constexpr int kB  = 4;
static constexpr int kNQ = 1024;
static constexpr int kNK = 2048;
static constexpr int kC  = 1024;
static constexpr int kH  = 16;
static constexpr int kD  = 64;

// ---- CDNA5 async copy: 16B global -> LDS (ASYNCcnt-tracked) ---------------
__device__ inline void async_copy16(const _Float16* g, _Float16* l) {
  // LDS byte address = low 32 bits of the flat shared-aperture address.
  uint32_t loff = (uint32_t)(uintptr_t)l;
  asm volatile("global_load_async_to_lds_b128 %0, %1, off"
               :: "v"(loff), "v"((uint64_t)(uintptr_t)g)
               : "memory");
}
__device__ inline void wait_async() {
  asm volatile("s_wait_asynccnt 0" ::: "memory");
}

// ---- fragment loaders (A: 16B vector loads, B: strided scalar loads) --------------------
__device__ inline v16h join16(v8h lo, v8h hi) {
  v16h r;
#pragma unroll
  for (int i = 0; i < 8; ++i) { r[i] = lo[i]; r[i + 8] = hi[i]; }
  return r;
}
// A fragment: row-major source, `row` already at this lane's row + k0.
__device__ inline v16h fragA(const _Float16* row, int lg) {
  const _Float16* p = row + 8 * lg;
  return join16(*(const v8h*)p, *(const v8h*)(p + 16));
}
// B fragment from a K-major ([k][n]) tile: column n0+lid, K in A's order
// (8*lg+i, 16+8*lg+i); caller passes &tile[n0 + lid], row stride ld.
__device__ inline v16h fragB(const _Float16* p, int ld, int lg) {
  v16h r;
#pragma unroll
  for (int i = 0; i < 8; ++i) {
    r[i] = p[(8 * lg + i) * ld];
    r[i + 8] = p[(16 + 8 * lg + i) * ld];
  }
  return r;
}

// ===========================================================================
// GEMM: out[M,N] = act[M,K] @ W[N,K]^T + bias.  Block 256 thr = 8 waves,
// block tile 64(M) x 256(N), K-step 32; wave tile 32x64 (8 WMMA / step).
// W tile staged TRANSPOSED [k][n] in LDS.
// ===========================================================================
static constexpr int LDA = 40;   // 32 + 8 pad halves (80B rows, 16B aligned)
static constexpr int LDW = 264;  // 256 + 8 pad halves (528B rows, 16B aligned)

template <bool ACT_F16, bool OUT_F16>
__device__ inline void gemm_body(const void* __restrict__ actv,
                                 const float* __restrict__ W,
                                 const float* __restrict__ bias,
                                 void* __restrict__ outv, int N, int K) {
  __shared__ _Float16 lds_a[64 * LDA];  // [m][k]
  __shared__ _Float16 lds_w[32 * LDW];  // [k][n]  (transposed W tile)

  const int t   = threadIdx.x;
  const int ln  = t & 31;
  const int wv  = t >> 5;
  const int lid = ln & 15;
  const int lg  = ln >> 4;
  const int mB  = blockIdx.y * 64;
  const int nB  = blockIdx.x * 256;
  const int mo  = (wv >> 2) * 32;    // two 16-row subtiles: mo, mo+16
  const int no  = (wv & 3) * 64;

  // staging roles: A: 64 rows x 4 k-groups of 8 ; W: thread t owns column n=t
  const int sm = t >> 2;          // 0..63
  const int sk = (t & 3) * 8;     // 0,8,16,24

  v8f acc[2][4] = {};
  for (int k0 = 0; k0 < K; k0 += 32) {
    __syncthreads();  // protect previous iteration's reads
    // ---- stage A tile (64 x 32) ----
    if (ACT_F16) {
      const _Float16* act = (const _Float16*)actv;
      *(v8h*)&lds_a[sm * LDA + sk] =
          *(const v8h*)(act + (size_t)(mB + sm) * K + k0 + sk);
    } else {
      const float* act = (const float*)actv;
      const float* ap = act + (size_t)(mB + sm) * K + k0 + sk;
      v4f a0 = *(const v4f*)ap;
      v4f a1 = *(const v4f*)(ap + 4);
      v8h ah;
#pragma unroll
      for (int e = 0; e < 4; ++e) { ah[e] = (_Float16)a0[e]; ah[e + 4] = (_Float16)a1[e]; }
      *(v8h*)&lds_a[sm * LDA + sk] = ah;
    }
    // ---- stage W tile transposed (32 x 256) + prefetch next K-step ----
    {
      const float* wr = W + (size_t)(nB + t) * K + k0;
      if (k0 + 32 < K) __builtin_prefetch(wr + 32, 0, 1);  // global_prefetch_b8
#pragma unroll
      for (int q = 0; q < 8; ++q) {
        v4f w4 = *(const v4f*)(wr + 4 * q);
#pragma unroll
        for (int e = 0; e < 4; ++e)
          lds_w[(4 * q + e) * LDW + t] = (_Float16)w4[e];
      }
    }
    __syncthreads();
    // ---- compute: wave tile 32 x 64 ----
    v16h a0 = fragA(&lds_a[(mo + lid) * LDA], lg);
    v16h a1 = fragA(&lds_a[(mo + 16 + lid) * LDA], lg);
#pragma unroll
    for (int j = 0; j < 4; ++j) {
      v16h bf = fragB(&lds_w[no + 16 * j + lid], LDW, lg);
      acc[0][j] = WMMA_F16(a0, bf, acc[0][j]);
      acc[1][j] = WMMA_F16(a1, bf, acc[1][j]);
    }
  }
  // ---- epilogue ----
#pragma unroll
  for (int i = 0; i < 2; ++i) {
#pragma unroll
    for (int j = 0; j < 4; ++j) {
      const int col = nB + no + 16 * j + lid;
      const float bv = bias[col];
#pragma unroll
      for (int r = 0; r < 8; ++r) {
        const int row = mB + mo + 16 * i + r + 8 * lg;
        if (OUT_F16)
          ((_Float16*)outv)[(size_t)row * N + col] = (_Float16)(acc[i][j][r] + bv);
        else
          ((float*)outv)[(size_t)row * N + col] = acc[i][j][r] + bv;
      }
    }
  }
}

__global__ __launch_bounds__(256) void gemm_f32act_f16out(
    const float* __restrict__ act, const float* __restrict__ W,
    const float* __restrict__ bias, _Float16* __restrict__ out, int N, int K) {
  gemm_body<false, true>(act, W, bias, out, N, K);
}
__global__ __launch_bounds__(256) void gemm_f16act_f32out(
    const _Float16* __restrict__ act, const float* __restrict__ W,
    const float* __restrict__ bias, float* __restrict__ out, int N, int K) {
  gemm_body<true, false>(act, W, bias, out, N, K);
}

// ===========================================================================
// Flash attention.  Grid (NQ/128, H, B), block 256 = 8 waves; each wave owns
// a 16-query tile; the block SHARES K/V chunks staged in LDS.  V chunk is
// copied with async global->LDS b128.
// ===========================================================================
static constexpr int LDK = 40;  // K chunk stored [d][tok], 32 tok + pad

__global__ __launch_bounds__(256) void flash_attn(
    const _Float16* __restrict__ qh, const _Float16* __restrict__ kvh,
    const uint8_t* __restrict__ mask, _Float16* __restrict__ oh) {
  __shared__ _Float16 lds_k[64 * LDK];      // [d][tok]  (transposed K chunk)
  __shared__ _Float16 lds_v[32 * 64];       // [tok][d]
  __shared__ _Float16 lds_p[8 * 16 * 32];   // per-wave P staging

  const int t   = threadIdx.x;
  const int ln  = t & 31;
  const int wv  = t >> 5;
  const int lid = ln & 15;
  const int lg  = ln >> 4;
  const int h   = blockIdx.y;
  const int b   = blockIdx.z;
  const int qr0 = blockIdx.x * 128 + wv * 16;
  const int ld  = 2 * kC;
  const float scale = 0.125f;  // 64^-0.5

  // staging roles: 32 tokens x 8 d-groups of 8
  const int stok = t >> 3;
  const int sd   = (t & 7) * 8;

  const _Float16* qrow = qh + (size_t)(b * kNQ + qr0 + lid) * kC + h * kD;
  const v16h qa0 = fragA(qrow, lg);
  const v16h qa1 = fragA(qrow + 32, lg);

  const _Float16* kbase = kvh + (size_t)b * kNK * ld + h * kD;
  const _Float16* vbase = kbase + kC;
  const uint8_t*  mrow  = mask + (size_t)b * kNQ * kNK;
  _Float16* pbuf = lds_p + wv * 512;

  float mrun[8], lrun[8];
  v8f o[4] = {};
#pragma unroll
  for (int r = 0; r < 8; ++r) { mrun[r] = -3.0e38f; lrun[r] = 0.0f; }

  for (int kn0 = 0; kn0 < kNK; kn0 += 32) {
    __syncthreads();  // protect previous chunk's reads
    // ---- stage K chunk transposed: lds_k[d][tok] ----
    {
      v8h kk = *(const v8h*)(kbase + (size_t)(kn0 + stok) * ld + sd);
#pragma unroll
      for (int e = 0; e < 8; ++e) lds_k[(sd + e) * LDK + stok] = kk[e];
    }
    // ---- stage V chunk (pure copy) via async global->LDS b128 ----
    async_copy16(vbase + (size_t)(kn0 + stok) * ld + sd, &lds_v[stok * 64 + sd]);
    wait_async();
    __syncthreads();

    // ---- S = (Q @ K^T) * scale over 32 keys (two 16-col tiles) ----
    v8f s[2];
#pragma unroll
    for (int nt = 0; nt < 2; ++nt) {
      v8f sacc = {};
      v16h b0 = fragB(&lds_k[nt * 16 + lid], LDK, lg);             // d 0..31
      v16h b1 = fragB(&lds_k[32 * LDK + nt * 16 + lid], LDK, lg);  // d 32..63
      sacc = WMMA_F16(qa0, b0, sacc);
      sacc = WMMA_F16(qa1, b1, sacc);
      s[nt] = sacc;
    }
    // scale + mask
#pragma unroll
    for (int nt = 0; nt < 2; ++nt) {
      const int tok = kn0 + nt * 16 + lid;
#pragma unroll
      for (int r = 0; r < 8; ++r) {
        const int qr = qr0 + r + 8 * lg;
        float sv = s[nt][r] * scale;
        if (mrow[(size_t)qr * kNK + tok]) sv = -3.0e38f;
        s[nt][r] = sv;
      }
    }
    // ---- online softmax (row reductions within each 16-lane group) ----
#pragma unroll
    for (int r = 0; r < 8; ++r) {
      float v = fmaxf(s[0][r], s[1][r]);
#pragma unroll
      for (int off = 8; off >= 1; off >>= 1) v = fmaxf(v, __shfl_xor(v, off, 16));
      const float mnew = fmaxf(mrun[r], v);
      const float corr = __expf(mrun[r] - mnew);
      const float p0 = __expf(s[0][r] - mnew);
      const float p1 = __expf(s[1][r] - mnew);
      float rs = p0 + p1;
#pragma unroll
      for (int off = 8; off >= 1; off >>= 1) rs += __shfl_xor(rs, off, 16);
      lrun[r] = lrun[r] * corr + rs;
      mrun[r] = mnew;
#pragma unroll
      for (int j = 0; j < 4; ++j) o[j][r] *= corr;
      const int prow = r + 8 * lg;
      pbuf[prow * 32 + lid]      = (_Float16)p0;
      pbuf[prow * 32 + 16 + lid] = (_Float16)p1;
    }
    asm volatile("s_wait_dscnt 0" ::: "memory");  // per-wave DS ordering

    // ---- O += P @ V ----
    const v16h pa = fragA(&pbuf[lid * 32], lg);
#pragma unroll
    for (int j = 0; j < 4; ++j) {
      const v16h vf = fragB(&lds_v[16 * j + lid], 64, lg);  // lane = d column
      o[j] = WMMA_F16(pa, vf, o[j]);
    }
  }

  // ---- epilogue: normalize, store as (b*NQ+q)*C + h*D + d ----
#pragma unroll
  for (int j = 0; j < 4; ++j) {
#pragma unroll
    for (int r = 0; r < 8; ++r) {
      const int qr = qr0 + r + 8 * lg;
      const float val = o[j][r] / lrun[r];
      oh[(size_t)(b * kNQ + qr) * kC + h * kD + 16 * j + lid] = (_Float16)val;
    }
  }
}

// ---------------------------------------------------------------------------
extern "C" void kernel_launch(void* const* d_in, const int* in_sizes, int n_in,
                              void* d_out, int out_size, void* d_ws, size_t ws_size,
                              hipStream_t stream) {
  const float*   x    = (const float*)d_in[0];
  const float*   y    = (const float*)d_in[1];
  const uint8_t* mask = (const uint8_t*)d_in[2];
  const float*   Wq   = (const float*)d_in[3];
  const float*   bq   = (const float*)d_in[4];
  const float*   Wkv  = (const float*)d_in[5];
  const float*   bkv  = (const float*)d_in[6];
  const float*   Wp   = (const float*)d_in[7];
  const float*   bp   = (const float*)d_in[8];
  float*         out  = (float*)d_out;

  // Workspace (fp16): q 8MB | kv 32MB | attn-out 8MB
  char* ws = (char*)d_ws;
  _Float16* qhb  = (_Float16*)(ws);
  _Float16* kvhb = (_Float16*)(ws + (size_t)(kB * kNQ) * kC * 2);
  _Float16* ohb  = (_Float16*)(ws + (size_t)(kB * kNQ) * kC * 2 +
                               (size_t)(kB * kNK) * (2 * kC) * 2);

  const dim3 blk(256);
  gemm_f32act_f16out<<<dim3(kC / 256, (kB * kNQ) / 64), blk, 0, stream>>>(
      x, Wq, bq, qhb, kC, kC);
  gemm_f32act_f16out<<<dim3((2 * kC) / 256, (kB * kNK) / 64), blk, 0, stream>>>(
      y, Wkv, bkv, kvhb, 2 * kC, kC);
  flash_attn<<<dim3(kNQ / 128, kH, kB), blk, 0, stream>>>(qhb, kvhb, mask, ohb);
  gemm_f16act_f32out<<<dim3(kC / 256, (kB * kNQ) / 64), blk, 0, stream>>>(
      ohb, Wp, bp, out, kC, kC);
}
